// SoftPointwiseKNN_27762668601761
// MI455X (gfx1250) — matrix operand order repaired and checked by reading, not compiled, not run on hardware
//
#include <hip/hip_runtime.h>
#include <hip/hip_bf16.h>
#include <math.h>

// ---------------------------------------------------------------------------
// SoftPointwiseKNN for MI455X (gfx1250, wave32, WMMA)
//   pass 1: per-row params (sigmoid head), row squared norms, min/max init
//   pass 1b (if ws large enough): pack x / x_n into bf16 hi/lo planes (split
//            precision: f32 ~= hi + lo), so the GEMM loop has zero conversion
//   pass 2: split-bf16 WMMA GEMM -> sim = -sqrt(max(x2+xn2-2*dot,0)) -> d_out
//           + per-row min/max via shfl reduction + ordered-uint atomics
//   pass 3: row min-max normalize + soft-KNN shift, in-place on d_out
// ---------------------------------------------------------------------------

#define NROWS 8192
#define MCOLS 8192
#define DDIM  256
#define EPSV  1e-12f

typedef __attribute__((ext_vector_type(16))) __bf16 v16bf;
typedef __attribute__((ext_vector_type(8)))  __bf16 bf16x8;
typedef __attribute__((ext_vector_type(4)))  __bf16 bf16x4;
typedef __attribute__((ext_vector_type(8)))  float  v8f;

union FragU { v16bf v; struct { bf16x8 lo8, hi8; } h; };

__device__ __forceinline__ unsigned f2ord(float f) {
    unsigned u = __float_as_uint(f);
    return (f >= 0.0f) ? (u | 0x80000000u) : ~u;
}
__device__ __forceinline__ float ord2f(unsigned u) {
    return (u & 0x80000000u) ? __uint_as_float(u & 0x7FFFFFFFu)
                             : __uint_as_float(~u);
}

// ---------------------------------------------------------------------------
// Pass 1: one wave (32 lanes) per row.
// ---------------------------------------------------------------------------
__global__ __launch_bounds__(256)
void prep_kernel(const float* __restrict__ x, const float* __restrict__ xn,
                 const float* __restrict__ W, const float* __restrict__ b,
                 float* __restrict__ kp, float* __restrict__ tp,
                 float* __restrict__ x2, float* __restrict__ xn2,
                 unsigned* __restrict__ mnE, unsigned* __restrict__ mxE)
{
    const int lane = threadIdx.x & 31;
    const int wave = threadIdx.x >> 5;
    const int row  = blockIdx.x * 8 + wave;

    if (row < NROWS) {
        const float* p = x + (size_t)row * DDIM;
        float s2 = 0.0f, d0 = 0.0f, d1 = 0.0f;
        #pragma unroll
        for (int c = lane; c < DDIM; c += 32) {
            float v = p[c];
            s2 += v * v;
            d0 += v * W[2 * c + 0];
            d1 += v * W[2 * c + 1];
        }
        #pragma unroll
        for (int m = 16; m >= 1; m >>= 1) {
            s2 += __shfl_xor(s2, m, 32);
            d0 += __shfl_xor(d0, m, 32);
            d1 += __shfl_xor(d1, m, 32);
        }
        if (lane == 0) {
            float kk = 1.0f / (1.0f + expf(-(d0 + b[0])));
            float tt = 1.0f / (1.0f + expf(-(d1 + b[1])));
            kk = fminf(fmaxf(kk, EPSV), 1.0f - EPSV);
            tt = fminf(fmaxf(tt, EPSV), 1.0f - EPSV);
            kp[row] = kk;
            tp[row] = tt;
            x2[row] = s2;
            mnE[row] = 0xFFFFFFFFu;   // encoded +inf  (min identity)
            mxE[row] = 0x00000000u;   // encoded -inf  (max identity)
        }
    } else {
        const int r = row - NROWS;
        const float* p = xn + (size_t)r * DDIM;
        float s2 = 0.0f;
        #pragma unroll
        for (int c = lane; c < DDIM; c += 32) {
            float v = p[c];
            s2 += v * v;
        }
        #pragma unroll
        for (int m = 16; m >= 1; m >>= 1) s2 += __shfl_xor(s2, m, 32);
        if (lane == 0) xn2[r] = s2;
    }
}

// ---------------------------------------------------------------------------
// Pass 1b: pack f32 -> split bf16 hi/lo planes (row-major [row][256]).
// ---------------------------------------------------------------------------
__global__ __launch_bounds__(256)
void pack_kernel(const float* __restrict__ x, const float* __restrict__ xn,
                 __bf16* __restrict__ xhi, __bf16* __restrict__ xlo,
                 __bf16* __restrict__ nhi, __bf16* __restrict__ nlo)
{
    const size_t TOT = (size_t)NROWS * DDIM;
    const size_t idx = ((size_t)blockIdx.x * blockDim.x + threadIdx.x) * 4;

    const float* src; __bf16* dh; __bf16* dl; size_t off;
    if (idx < TOT) { src = x;  dh = xhi; dl = xlo; off = idx; }
    else           { src = xn; dh = nhi; dl = nlo; off = idx - TOT; }

    float4 v = *(const float4*)(src + off);
    float f[4] = {v.x, v.y, v.z, v.w};
    bf16x4 h, l;
    #pragma unroll
    for (int e = 0; e < 4; ++e) {
        __bf16 hh = (__bf16)f[e];
        h[e] = hh;
        l[e] = (__bf16)(f[e] - (float)hh);
    }
    *(bf16x4*)(dh + off) = h;
    *(bf16x4*)(dl + off) = l;
}

// ---------------------------------------------------------------------------
// Shared epilogue: sim = -sqrt(max(x2+xn2-2*dot,0)), store + row min/max.
// Wave tile: rows [rBase,rBase+32), cols [cBase,cBase+64).
// ---------------------------------------------------------------------------
__device__ __forceinline__
void sim_epilogue(const v8f acc[2][4], int rBase, int cBase, int hi, int ln,
                  const float* __restrict__ x2, const float* __restrict__ xn2,
                  unsigned* __restrict__ mnE, unsigned* __restrict__ mxE,
                  float* __restrict__ simOut)
{
    float xn2c[4];
    #pragma unroll
    for (int nt = 0; nt < 4; ++nt) xn2c[nt] = xn2[cBase + nt * 16 + ln];

    #pragma unroll
    for (int mt = 0; mt < 2; ++mt) {
        float rmn[8], rmx[8], x2r[8];
        int   rown[8];
        #pragma unroll
        for (int j = 0; j < 8; ++j) {
            rown[j] = rBase + mt * 16 + hi * 8 + j;  // C layout: half picks M 0-7/8-15
            x2r[j]  = x2[rown[j]];
            rmn[j]  = __builtin_huge_valf();
            rmx[j]  = -__builtin_huge_valf();
        }
        #pragma unroll
        for (int nt = 0; nt < 4; ++nt) {
            const int col = cBase + nt * 16 + ln;
            #pragma unroll
            for (int j = 0; j < 8; ++j) {
                float d2 = fmaxf(x2r[j] + xn2c[nt] - 2.0f * acc[mt][nt][j], 0.0f);
                float s  = -sqrtf(d2);
                simOut[(size_t)rown[j] * MCOLS + col] = s;
                rmn[j] = fminf(rmn[j], s);
                rmx[j] = fmaxf(rmx[j], s);
            }
        }
        #pragma unroll
        for (int m = 8; m >= 1; m >>= 1) {   // xor<16 stays within the lane half
            #pragma unroll
            for (int j = 0; j < 8; ++j) {
                rmn[j] = fminf(rmn[j], __shfl_xor(rmn[j], m, 32));
                rmx[j] = fmaxf(rmx[j], __shfl_xor(rmx[j], m, 32));
            }
        }
        if (ln == 0) {
            #pragma unroll
            for (int j = 0; j < 8; ++j) {
                atomicMin(&mnE[rown[j]], f2ord(rmn[j]));
                atomicMax(&mxE[rown[j]], f2ord(rmx[j]));
            }
        }
    }
}

// ---------------------------------------------------------------------------
// Pass 2 (fast path): pre-packed bf16 hi/lo, pure load+WMMA inner loop.
// 128x128 tile per workgroup; 8 waves, each wave 32(M) x 64(N).
// ---------------------------------------------------------------------------
__global__ __launch_bounds__(256)
void gemm_sim_packed_kernel(const __bf16* __restrict__ xhi, const __bf16* __restrict__ xlo,
                            const __bf16* __restrict__ nhi, const __bf16* __restrict__ nlo,
                            const float* __restrict__ x2, const float* __restrict__ xn2,
                            unsigned* __restrict__ mnE, unsigned* __restrict__ mxE,
                            float* __restrict__ simOut)
{
    const int lane = threadIdx.x & 31;
    const int wave = threadIdx.x >> 5;
    const int hi   = lane >> 4;
    const int ln   = lane & 15;

    const int rBase = blockIdx.y * 128 + (wave >> 1) * 32;
    const int cBase = blockIdx.x * 128 + (wave & 1) * 64;

    v8f acc[2][4];
    #pragma unroll
    for (int mt = 0; mt < 2; ++mt)
        #pragma unroll
        for (int nt = 0; nt < 4; ++nt)
            #pragma unroll
            for (int e = 0; e < 8; ++e) acc[mt][nt][e] = 0.0f;

    // A 16x32 bf16 fragment: half h -> K = h*16 + {0..15} (contiguous)
    const int aoff = hi * 16;
    // B 32x16 bf16 fragment: half h -> K = h*16 + {0..15} (contiguous)
    const int boff = hi * 16;

    for (int k0 = 0; k0 < DDIM; k0 += 32) {
        v16bf ah[2], al[2], bh[4], bl[4];

        #pragma unroll
        for (int mt = 0; mt < 2; ++mt) {
            const size_t base = (size_t)(rBase + mt * 16 + ln) * DDIM + k0 + aoff;
            FragU f, g;
            f.h.lo8 = *(const bf16x8*)(xhi + base);
            f.h.hi8 = *(const bf16x8*)(xhi + base + 8);
            g.h.lo8 = *(const bf16x8*)(xlo + base);
            g.h.hi8 = *(const bf16x8*)(xlo + base + 8);
            ah[mt] = f.v;
            al[mt] = g.v;
        }
        #pragma unroll
        for (int nt = 0; nt < 4; ++nt) {
            const size_t base = (size_t)(cBase + nt * 16 + ln) * DDIM + k0 + boff;
            FragU f, g;
            f.h.lo8 = *(const bf16x8*)(nhi + base);
            f.h.hi8 = *(const bf16x8*)(nhi + base + 8);
            g.h.lo8 = *(const bf16x8*)(nlo + base);
            g.h.hi8 = *(const bf16x8*)(nlo + base + 8);
            bh[nt] = f.v;
            bl[nt] = g.v;
        }

        // term-major issue order: consecutive WMMAs hit different accumulators
        #pragma unroll
        for (int term = 0; term < 3; ++term)
            #pragma unroll
            for (int mt = 0; mt < 2; ++mt)
                #pragma unroll
                for (int nt = 0; nt < 4; ++nt) {
                    const v16bf& A = (term == 2) ? al[mt] : ah[mt];
                    const v16bf& B = (term == 1) ? bl[nt] : bh[nt];
                    acc[mt][nt] = __builtin_amdgcn_wmma_f32_16x16x32_bf16(
                        false, A, false, B, (short)0, acc[mt][nt], false, false);
                }
    }

    sim_epilogue(acc, rBase, cBase, hi, ln, x2, xn2, mnE, mxE, simOut);
}

// ---------------------------------------------------------------------------
// Pass 2 (fallback): convert f32 -> split bf16 inside the loop (used only if
// ws is too small for the packed planes).
// ---------------------------------------------------------------------------
__global__ __launch_bounds__(256)
void gemm_sim_kernel(const float* __restrict__ x, const float* __restrict__ xn,
                     const float* __restrict__ x2, const float* __restrict__ xn2,
                     unsigned* __restrict__ mnE, unsigned* __restrict__ mxE,
                     float* __restrict__ simOut)
{
    const int lane = threadIdx.x & 31;
    const int wave = threadIdx.x >> 5;
    const int hi   = lane >> 4;
    const int ln   = lane & 15;

    const int rBase = blockIdx.y * 128 + (wave >> 1) * 32;
    const int cBase = blockIdx.x * 128 + (wave & 1) * 64;

    v8f acc[2][4];
    #pragma unroll
    for (int mt = 0; mt < 2; ++mt)
        #pragma unroll
        for (int nt = 0; nt < 4; ++nt)
            #pragma unroll
            for (int e = 0; e < 8; ++e) acc[mt][nt][e] = 0.0f;

    const int aoff = hi * 16;
    const int boff = hi * 16;

    for (int k0 = 0; k0 < DDIM; k0 += 32) {
        v16bf ah[2], al[2], bh[4], bl[4];

        #pragma unroll
        for (int mt = 0; mt < 2; ++mt) {
            const float* p = x + (size_t)(rBase + mt * 16 + ln) * DDIM + k0 + aoff;
            float v[16];
            *(float4*)(v + 0)  = *(const float4*)(p + 0);
            *(float4*)(v + 4)  = *(const float4*)(p + 4);
            *(float4*)(v + 8)  = *(const float4*)(p + 8);
            *(float4*)(v + 12) = *(const float4*)(p + 12);
            #pragma unroll
            for (int e = 0; e < 16; ++e) {
                __bf16 h = (__bf16)v[e];
                ah[mt][e] = h;
                al[mt][e] = (__bf16)(v[e] - (float)h);
            }
        }
        #pragma unroll
        for (int nt = 0; nt < 4; ++nt) {
            const float* p = xn + (size_t)(cBase + nt * 16 + ln) * DDIM + k0 + boff;
            float v[16];
            *(float4*)(v + 0)  = *(const float4*)(p + 0);
            *(float4*)(v + 4)  = *(const float4*)(p + 4);
            *(float4*)(v + 8)  = *(const float4*)(p + 8);
            *(float4*)(v + 12) = *(const float4*)(p + 12);
            #pragma unroll
            for (int e = 0; e < 16; ++e) {
                __bf16 h = (__bf16)v[e];
                bh[nt][e] = h;
                bl[nt][e] = (__bf16)(v[e] - (float)h);
            }
        }

        #pragma unroll
        for (int term = 0; term < 3; ++term)
            #pragma unroll
            for (int mt = 0; mt < 2; ++mt)
                #pragma unroll
                for (int nt = 0; nt < 4; ++nt) {
                    const v16bf& A = (term == 2) ? al[mt] : ah[mt];
                    const v16bf& B = (term == 1) ? bl[nt] : bh[nt];
                    acc[mt][nt] = __builtin_amdgcn_wmma_f32_16x16x32_bf16(
                        false, A, false, B, (short)0, acc[mt][nt], false, false);
                }
    }

    sim_epilogue(acc, rBase, cBase, hi, ln, x2, xn2, mnE, mxE, simOut);
}

// ---------------------------------------------------------------------------
// Pass 3: in-place min-max normalize + soft-KNN shift (float4 vectorized)
// ---------------------------------------------------------------------------
__global__ __launch_bounds__(256)
void finalize_kernel(float* __restrict__ out,
                     const float* __restrict__ kp, const float* __restrict__ tp,
                     const unsigned* __restrict__ mnE, const unsigned* __restrict__ mxE)
{
    const size_t tid = (size_t)blockIdx.x * blockDim.x + threadIdx.x;
    const size_t idx = tid * 4;
    const int row = (int)(idx >> 13);   // MCOLS == 8192 == 2^13

    float4 v = *(float4*)(out + idx);

    const float mn = ord2f(mnE[row]);
    const float mx = ord2f(mxE[row]);
    const float inv = 1.0f / (mx - mn);
    const float k = kp[row];
    const float t = tp[row];
    const float tt = t / (1.0f - t);
    const float logk = logf(k);

    float s[4] = {v.x, v.y, v.z, v.w};
    #pragma unroll
    for (int e = 0; e < 4; ++e) {
        float si = fmaxf((s[e] - mn) * inv, EPSV);
        float shift = tt * (k - si) * (logf(si) - logk);
        s[e] = si + ((si <= k) ? shift : 0.0f);
    }
    v.x = s[0]; v.y = s[1]; v.z = s[2]; v.w = s[3];
    *(float4*)(out + idx) = v;
}

// ---------------------------------------------------------------------------
extern "C" void kernel_launch(void* const* d_in, const int* in_sizes, int n_in,
                              void* d_out, int out_size, void* d_ws, size_t ws_size,
                              hipStream_t stream)
{
    const float* x  = (const float*)d_in[0];
    const float* xn = (const float*)d_in[1];
    const float* W  = (const float*)d_in[2];
    const float* b  = (const float*)d_in[3];
    float* out = (float*)d_out;

    float* ws = (float*)d_ws;
    float* kp  = ws;                 // N
    float* tp  = ws + NROWS;         // N
    float* x2  = ws + 2 * NROWS;     // N
    float* xn2 = ws + 3 * NROWS;     // M
    unsigned* mnE = (unsigned*)(ws + 3 * NROWS + MCOLS);  // N
    unsigned* mxE = mnE + NROWS;                          // N

    const size_t statsBytes = (size_t)6 * NROWS * sizeof(float);      // 192 KB
    const size_t TOT        = (size_t)NROWS * DDIM;                   // elems/plane
    const size_t packBytes  = 4 * TOT * sizeof(__bf16);               // 16 MB

    // pass 1
    prep_kernel<<<(NROWS + MCOLS) / 8, 256, 0, stream>>>(
        x, xn, W, b, kp, tp, x2, xn2, mnE, mxE);

    dim3 g2(MCOLS / 128, NROWS / 128);

    if (ws_size >= statsBytes + packBytes) {
        __bf16* xhi = (__bf16*)((char*)d_ws + statsBytes);
        __bf16* xlo = xhi + TOT;
        __bf16* nhi = xlo + TOT;
        __bf16* nlo = nhi + TOT;

        // pass 1b: pack both matrices (2*TOT elements, 4 per thread)
        pack_kernel<<<(unsigned)(2 * TOT / 4 / 256), 256, 0, stream>>>(
            x, xn, xhi, xlo, nhi, nlo);

        gemm_sim_packed_kernel<<<g2, 256, 0, stream>>>(
            xhi, xlo, nhi, nlo, x2, xn2, mnE, mxE, out);
    } else {
        gemm_sim_kernel<<<g2, 256, 0, stream>>>(
            x, xn, x2, xn2, mnE, mxE, out);
    }

    // pass 3
    const size_t total4 = (size_t)NROWS * MCOLS / 4;
    finalize_kernel<<<(unsigned)(total4 / 256), 256, 0, stream>>>(
        out, kp, tp, mnE, mxE);
}
